// RWKVTimeMixing_9234179686554
// MI455X (gfx1250) — matrix operand order repaired and checked by reading, not compiled, not run on hardware
//
#include <hip/hip_runtime.h>
#include <hip/hip_bf16.h>

// ---------------------------------------------------------------- constants
#define B_  4
#define S_  2048
#define D_  1024
#define H_  16
#define HD_ 64
#define M_  (B_ * S_)          // 8192 rows of activations

typedef __attribute__((ext_vector_type(16))) __bf16 v16bf;
typedef __attribute__((ext_vector_type(8)))  float  v8f;

// fp32 -> bf16 round-to-nearest-even
static __device__ __forceinline__ unsigned short f2bf(float f) {
    unsigned int u = __float_as_uint(f);
    u += 0x7FFFu + ((u >> 16) & 1u);
    return (unsigned short)(u >> 16);
}

// ---- gfx1250 async global->LDS copy (ASYNCcnt-tracked), via inline asm since
// ---- the clang builtin arity differs across toolchains (see CDNA5_HIP.md).
static __device__ __forceinline__ void async_copy_b128(void* lds, const void* gaddr) {
    unsigned lds_off = (unsigned)(size_t)lds;   // generic->LDS: HW truncates to LDS offset
    asm volatile("global_load_async_to_lds_b128 %0, %1, off"
                 :: "v"(lds_off), "v"(gaddr) : "memory");
}
#define WAIT_ASYNC_0() asm volatile("s_wait_asynccnt 0x0" ::: "memory")
#define WAIT_ASYNC_4() asm volatile("s_wait_asynccnt 0x4" ::: "memory")

// ---------------------------------------------------------------- kernel 1: fp32 -> bf16
__global__ __launch_bounds__(256) void cvt_bf16_kernel(const float* __restrict__ src,
                                                       unsigned short* __restrict__ dst, int n) {
    int i = blockIdx.x * 256 + threadIdx.x;
    if (i < n) dst[i] = f2bf(src[i]);
}

// ---------------------------------------------------------------- kernel 2: token shift + mix
__global__ __launch_bounds__(256) void mix_kernel(const float* __restrict__ x,
                                                  const float* __restrict__ tmr,
                                                  const float* __restrict__ tmk,
                                                  const float* __restrict__ tmv,
                                                  unsigned short* __restrict__ xr,
                                                  unsigned short* __restrict__ xk,
                                                  unsigned short* __restrict__ xv,
                                                  int total) {
    int i = blockIdx.x * 256 + threadIdx.x;
    if (i >= total) return;
    int d  = i % D_;
    int bs = i / D_;
    int s  = bs % S_;
    float xc = x[i];
    float xp = (s == 0) ? 0.0f : x[i - D_];
    float mr = tmr[d], mk = tmk[d], mv = tmv[d];
    xr[i] = f2bf(xc * mr + xp * (1.0f - mr));
    xk[i] = f2bf(xc * mk + xp * (1.0f - mk));
    xv[i] = f2bf(xc * mv + xp * (1.0f - mv));
}

// ---------------------------------------------------------------- WMMA GEMM
//  C[M,N] (f32) = A[M,K] (bf16 row-major) @ W[N,K]^T (bf16 row-major)
//  256 threads = 8 waves (4M x 2N); block tile 128x128; wave tile 32x64
//  (2 M-frags x 4 N-frags = 8 wmma per K-chunk); K chunked by 32, LDS
//  double-buffered and filled with async global->LDS B128 copies.
#define BM 128
#define BN 128
#define BK 32
#define LDA 40   // ushort row stride: 80 B = multiple of 16 B (b128-aligned)
#define LDB 40

union FragU { unsigned int u[8]; uint4 q[2]; v16bf v; };

__global__ __launch_bounds__(256) void gemm_bf16_wmma(const unsigned short* __restrict__ A,
                                                      const unsigned short* __restrict__ W,
                                                      float* __restrict__ C,
                                                      int M, int N, int K) {
    __shared__ unsigned short sA[2][BM * LDA];
    __shared__ unsigned short sB[2][BN * LDB];

    const int tid  = threadIdx.x;
    const int lane = tid & 31;
    const int wave = tid >> 5;
    const int wr   = wave >> 1;            // 0..3: 32-row M group
    const int wc   = wave & 1;             // 0..1: 64-col N group
    const int hlf  = lane >> 4;            // lane half
    const int l15  = lane & 15;
    const int n0   = blockIdx.x * BN;
    const int m0   = blockIdx.y * BM;

    v8f acc[2][4];
#pragma unroll
    for (int mt = 0; mt < 2; ++mt)
#pragma unroll
        for (int nt = 0; nt < 4; ++nt)
            acc[mt][nt] = (v8f){0.f, 0.f, 0.f, 0.f, 0.f, 0.f, 0.f, 0.f};

    // staging: 128 rows x 64 B per matrix = 512 x b128; 2 chunks/thread/matrix.
    // Per-thread global pointers advanced by BK each chunk (no re-derivation).
    const int c0   = tid;
    const int c1   = tid + 256;
    const int row0 = c0 >> 2, seg0 = (c0 & 3) * 8;   // seg in ushorts (16 B)
    const int row1 = c1 >> 2, seg1 = (c1 & 3) * 8;

    const unsigned short* gA0 = A + (size_t)(m0 + row0) * K + seg0;
    const unsigned short* gA1 = A + (size_t)(m0 + row1) * K + seg1;
    const unsigned short* gB0 = W + (size_t)(n0 + row0) * K + seg0;
    const unsigned short* gB1 = W + (size_t)(n0 + row1) * K + seg1;

    auto stage = [&](int buf) {
        async_copy_b128(&sA[buf][row0 * LDA + seg0], gA0);
        async_copy_b128(&sB[buf][row0 * LDB + seg0], gB0);
        async_copy_b128(&sA[buf][row1 * LDA + seg1], gA1);
        async_copy_b128(&sB[buf][row1 * LDB + seg1], gB1);
        gA0 += BK; gA1 += BK; gB0 += BK; gB1 += BK;
    };

    stage(0);
    int buf = 0;
    for (int kk = 0; kk < K; kk += BK) {
        const bool has_next = (kk + BK < K);
        if (has_next) stage(buf ^ 1);   // overlap next chunk's HBM traffic with compute
        if (has_next) { WAIT_ASYNC_4(); } else { WAIT_ASYNC_0(); }
        __syncthreads();

        // Preload ALL fragments first so dscnt waits pipeline (no per-wmma stall).
        // A fragments: per lane two contiguous 16 B runs (ISA 16-bit A layout).
        FragU aF[2];
#pragma unroll
        for (int mt = 0; mt < 2; ++mt) {
            const unsigned short* rowp = &sA[buf][(wr * 32 + mt * 16 + l15) * LDA];
            aF[mt].q[0] = *(const uint4*)(rowp + hlf * 8);        // K 0..7  (half-sel)
            aF[mt].q[1] = *(const uint4*)(rowp + 16 + hlf * 8);   // K 16..23 (half-sel)
        }
        // B fragments: per lane two contiguous 16 B runs (same K order as A).
        FragU bF[4];
#pragma unroll
        for (int nt = 0; nt < 4; ++nt) {
            const unsigned short* colp = &sB[buf][(wc * 64 + nt * 16 + l15) * LDB];
            bF[nt].q[0] = *(const uint4*)(colp + hlf * 8);
            bF[nt].q[1] = *(const uint4*)(colp + 16 + hlf * 8);
        }
        // 8 back-to-back WMMAs on the matrix pipe.
#pragma unroll
        for (int nt = 0; nt < 4; ++nt)
#pragma unroll
            for (int mt = 0; mt < 2; ++mt)
                acc[mt][nt] = __builtin_amdgcn_wmma_f32_16x16x32_bf16(
                    false, aF[mt].v, false, bF[nt].v, (short)0, acc[mt][nt], false, false);

        __syncthreads();     // all waves done with buf before it is re-staged
        buf ^= 1;
    }

    // epilogue: C/D layout: VGPR i -> M = i + hlf*8 ; lane&15 -> N
#pragma unroll
    for (int mt = 0; mt < 2; ++mt) {
        int mbase = m0 + wr * 32 + mt * 16 + hlf * 8;
        int nb    = n0 + wc * 64 + l15;
#pragma unroll
        for (int nt = 0; nt < 4; ++nt)
#pragma unroll
            for (int i = 0; i < 8; ++i)
                C[(size_t)(mbase + i) * N + nb + nt * 16] = acc[mt][nt][i];
    }
}

// ---------------------------------------------------------------- kernel 4: WKV scan
__global__ __launch_bounds__(256) void wkv_scan_kernel(const float* __restrict__ k,
                                                       const float* __restrict__ v,
                                                       const float* __restrict__ time_decay,
                                                       const float* __restrict__ time_first,
                                                       float* __restrict__ wkv) {
    int c = blockIdx.x * 256 + threadIdx.x;        // 0 .. B*D-1
    if (c >= B_ * D_) return;
    int b  = c / D_;
    int ch = c % D_;

    float w = -__expf(time_decay[ch]);
    float u = time_first[ch];

    const float* kp = k   + (size_t)b * S_ * D_ + ch;
    const float* vp = v   + (size_t)b * S_ * D_ + ch;
    float*       op = wkv + (size_t)b * S_ * D_ + ch;

    float a   = 0.0f;
    float bst = -1e38f;
    for (int t = 0; t < S_; ++t) {
        size_t off = (size_t)t * D_;
        float kt = kp[off];
        float vt = vp[off];
        if (t + 1 < S_) {
            __builtin_prefetch(&kp[off + D_], 0, 1);
            __builtin_prefetch(&vp[off + D_], 0, 1);
        }
        float wk = kt + u;
        float p  = fmaxf(bst, wk);
        float e1 = __expf(bst - p);
        float e2 = __expf(wk - p);
        op[off]  = (e1 * a + e2 * vt) / (e1 + e2 + 1e-8f);

        float ww  = bst + w;
        float p2  = fmaxf(ww, kt);
        float e1b = __expf(ww - p2);
        float e2b = __expf(kt - p2);
        a   = e1b * a + e2b * vt;
        bst = p2 + __logf(e1b + e2b + 1e-8f);
    }
}

// ---------------------------------------------------------------- kernel 5: GroupNorm stats
__global__ __launch_bounds__(256) void gn_stats_kernel(const float* __restrict__ r,
                                                       const float* __restrict__ wkv,
                                                       float* __restrict__ stats) {
    int bh = blockIdx.x;               // 0 .. B*H-1
    int b  = bh / H_;
    int hh = bh % H_;
    int tid = threadIdx.x;

    float sum = 0.f, ss = 0.f;
    const int cnt = S_ * HD_;
    for (int i = tid; i < cnt; i += 256) {
        int t = i / HD_;
        int e = i % HD_;
        size_t idx = ((size_t)(b * S_ + t)) * D_ + hh * HD_ + e;
        float rv = r[idx];
        float y  = wkv[idx] * (1.0f / (1.0f + __expf(-rv)));
        sum += y;
        ss  += y * y;
    }
    __shared__ float s1[256], s2[256];
    s1[tid] = sum; s2[tid] = ss;
    __syncthreads();
    for (int off = 128; off > 0; off >>= 1) {
        if (tid < off) { s1[tid] += s1[tid + off]; s2[tid] += s2[tid + off]; }
        __syncthreads();
    }
    if (tid == 0) {
        float inv  = 1.0f / (float)cnt;
        float mean = s1[0] * inv;
        float var  = s2[0] * inv - mean * mean;
        stats[2 * bh]     = mean;
        stats[2 * bh + 1] = rsqrtf(var + 1e-5f);
    }
}

// ---------------------------------------------------------------- kernel 6: gate + normalize -> bf16
__global__ __launch_bounds__(256) void gn_apply_kernel(const float* __restrict__ r,
                                                       const float* __restrict__ wkv,
                                                       const float* __restrict__ stats,
                                                       const float* __restrict__ gamma,
                                                       const float* __restrict__ beta,
                                                       unsigned short* __restrict__ normed,
                                                       int total) {
    int i = blockIdx.x * 256 + threadIdx.x;
    if (i >= total) return;
    int d  = i % D_;
    int bs = i / D_;
    int b  = bs / S_;
    int hh = d / HD_;
    int bh = b * H_ + hh;
    float rv = r[i];
    float y  = wkv[i] * (1.0f / (1.0f + __expf(-rv)));
    float g  = (y - stats[2 * bh]) * stats[2 * bh + 1] * gamma[d] + beta[d];
    normed[i] = f2bf(g);
}

// ---------------------------------------------------------------- launch
extern "C" void kernel_launch(void* const* d_in, const int* in_sizes, int n_in,
                              void* d_out, int out_size, void* d_ws, size_t ws_size,
                              hipStream_t stream) {
    const float* x      = (const float*)d_in[0];
    const float* tmr    = (const float*)d_in[1];
    const float* tmk    = (const float*)d_in[2];
    const float* tmv    = (const float*)d_in[3];
    const float* W_r    = (const float*)d_in[4];
    const float* W_k    = (const float*)d_in[5];
    const float* W_v    = (const float*)d_in[6];
    const float* W_o    = (const float*)d_in[7];
    const float* tdec   = (const float*)d_in[8];
    const float* tfirst = (const float*)d_in[9];
    const float* gamma  = (const float*)d_in[10];
    const float* beta   = (const float*)d_in[11];
    float* out = (float*)d_out;

    // workspace carving
    char* ws = (char*)d_ws;
    size_t off = 0;
    auto carve = [&](size_t bytes) -> char* {
        char* p = ws + off;
        off = (off + bytes + 255) & ~(size_t)255;
        return p;
    };
    const size_t actBf = (size_t)M_ * D_ * sizeof(unsigned short);
    const size_t wBf   = (size_t)D_ * D_ * sizeof(unsigned short);
    const size_t actF  = (size_t)M_ * D_ * sizeof(float);

    unsigned short* xr  = (unsigned short*)carve(actBf);
    unsigned short* xk  = (unsigned short*)carve(actBf);
    unsigned short* xv  = (unsigned short*)carve(actBf);
    unsigned short* wrb = (unsigned short*)carve(wBf);
    unsigned short* wkb = (unsigned short*)carve(wBf);
    unsigned short* wvb = (unsigned short*)carve(wBf);
    unsigned short* wob = (unsigned short*)carve(wBf);
    float* rbuf   = (float*)carve(actF);
    float* kbuf   = (float*)carve(actF);
    float* vbuf   = (float*)carve(actF);
    float* wkvbuf = (float*)carve(actF);
    unsigned short* normed = (unsigned short*)carve(actBf);
    float* stats  = (float*)carve((size_t)B_ * H_ * 2 * sizeof(float));
    (void)ws_size; (void)in_sizes; (void)n_in; (void)out_size;

    const int nW = D_ * D_;
    // 1) convert weights fp32 -> bf16
    cvt_bf16_kernel<<<nW / 256, 256, 0, stream>>>(W_r, wrb, nW);
    cvt_bf16_kernel<<<nW / 256, 256, 0, stream>>>(W_k, wkb, nW);
    cvt_bf16_kernel<<<nW / 256, 256, 0, stream>>>(W_v, wvb, nW);
    cvt_bf16_kernel<<<nW / 256, 256, 0, stream>>>(W_o, wob, nW);

    // 2) token shift + mix (fp32 -> bf16)
    const int total = M_ * D_;
    mix_kernel<<<(total + 255) / 256, 256, 0, stream>>>(x, tmr, tmk, tmv, xr, xk, xv, total);

    // 3) r/k/v projections: [8192,1024] = [8192,1024] @ [1024,1024]^T
    dim3 ggrid(D_ / BN, M_ / BM);   // (8, 64)
    dim3 gblk(256);
    gemm_bf16_wmma<<<ggrid, gblk, 0, stream>>>(xr, wrb, rbuf, M_, D_, D_);
    gemm_bf16_wmma<<<ggrid, gblk, 0, stream>>>(xk, wkb, kbuf, M_, D_, D_);
    gemm_bf16_wmma<<<ggrid, gblk, 0, stream>>>(xv, wvb, vbuf, M_, D_, D_);

    // 4) WKV sequential scan: one lane per (b, channel)
    wkv_scan_kernel<<<(B_ * D_) / 256, 256, 0, stream>>>(kbuf, vbuf, tdec, tfirst, wkvbuf);

    // 5) GroupNorm statistics per (b, h)
    gn_stats_kernel<<<B_ * H_, 256, 0, stream>>>(rbuf, wkvbuf, stats);

    // 6) gate + normalize -> bf16
    gn_apply_kernel<<<(total + 255) / 256, 256, 0, stream>>>(rbuf, wkvbuf, stats, gamma, beta,
                                                             normed, total);

    // 7) output projection -> d_out (fp32)
    gemm_bf16_wmma<<<ggrid, gblk, 0, stream>>>(normed, wob, out, M_, D_, D_);
}
